// OriginalImplementationFastAPLoss_19284403159483
// MI455X (gfx1250) — compile-verified, matrix operand order checked by reading, not run on hardware
//
#include <hip/hip_runtime.h>

// ---------------------------------------------------------------------------
// FastAP loss, MI455X (gfx1250, wave32, WMMA).
//   Phase 0: f32 -> f16 convert of embeddings (2 MB, L2-resident), zero accum.
//   Phase 1: tiled Gram via v_wmma_f32_16x16x32_f16 (software-pipelined B
//            fragments); analytic 2-bin soft histogram scatter into LDS
//            (ds_add_f32, branchless); one global write per row.
//   Phase 2: per-row cumsum/AP, global atomic reduction.
//   Phase 3: loss = 1 - sum_ap / n_valid.
// ---------------------------------------------------------------------------

#define N_SAMPLES 8192
#define EMBED     128
#define NBINS     10
#define LBINS     11          // NBINS + 1 bin centers
#define HL        12          // padded LDS histogram row length
#define HIST_WAVES 8          // 256 threads per workgroup

typedef __attribute__((ext_vector_type(16))) _Float16 half16;
typedef __attribute__((ext_vector_type(8)))  _Float16 half8;
typedef __attribute__((ext_vector_type(8)))  float    float8;

// ---------------- Phase 0: convert + zero accumulators ----------------------
__global__ void fastap_convert_kernel(const float* __restrict__ x,
                                      _Float16* __restrict__ xh,
                                      float* __restrict__ acc) {
  int i = blockIdx.x * blockDim.x + threadIdx.x;
  if (i < N_SAMPLES * EMBED) xh[i] = (_Float16)x[i];
  if (i < 2) acc[i] = 0.0f;
}

// Load one 16x32 f16 A/B fragment (per-lane K split per CDNA5 WMMA layout).
__device__ __forceinline__ half16 load_frag(const _Float16* __restrict__ rowp,
                                            int kc, int lanehalf) {
  half8 lo = *(const half8*)(rowp + kc + lanehalf * 8);
  half8 hi = *(const half8*)(rowp + kc + 16 + lanehalf * 8);
  half16 r;
#pragma unroll
  for (int i = 0; i < 8; ++i) { r[i] = lo[i]; r[i + 8] = hi[i]; }
  return r;
}

// ---------------- Phase 1: Gram tiles + soft-histogram scatter --------------
__global__ void __launch_bounds__(32 * HIST_WAVES)
fastap_hist_kernel(const _Float16* __restrict__ xh,
                   const int* __restrict__ labels,
                   float* __restrict__ g_hpos,
                   float* __restrict__ g_hneg) {
  __shared__ float s_h[2][16][HL];   // [pos/neg][row][bin]
  __shared__ int   s_rlab[16];

  const int tid      = threadIdx.x;
  const int wave     = tid >> 5;
  const int lane     = tid & 31;
  const int lanehalf = lane >> 4;    // 0: lanes 0-15, 1: lanes 16-31
  const int m        = lane & 15;    // row within A tile / col within C tile
  const int rowBase  = blockIdx.x * 16;

  for (int i = tid; i < 2 * 16 * HL; i += blockDim.x) (&s_h[0][0][0])[i] = 0.0f;
  if (tid < 16) s_rlab[tid] = labels[rowBase + tid];
  __syncthreads();

  // A fragments: this wave's 16 rows x full K=128, resident in VGPRs.
  const _Float16* __restrict__ arow = xh + (size_t)(rowBase + m) * EMBED;
  half16 A0 = load_frag(arow,  0, lanehalf);
  half16 A1 = load_frag(arow, 32, lanehalf);
  half16 A2 = load_frag(arow, 64, lanehalf);
  half16 A3 = load_frag(arow, 96, lanehalf);

  // Row labels this lane's C-fragment touches (rows lanehalf*8 + v).
  int rl[8];
#pragma unroll
  for (int v = 0; v < 8; ++v) rl[v] = s_rlab[lanehalf * 8 + v];

  const float invDelta = (float)NBINS / 4.0f;   // 1/Delta = 2.5
  const int   nColTiles = N_SAMPLES / 16;       // 512

  // -------- software-pipelined column-tile loop (double-buffered B) --------
  int ct = wave;
  int gcol = ct * 16 + m;
  const _Float16* brow = xh + (size_t)gcol * EMBED;
  half16 B0 = load_frag(brow,  0, lanehalf);
  half16 B1 = load_frag(brow, 32, lanehalf);
  half16 B2 = load_frag(brow, 64, lanehalf);
  half16 B3 = load_frag(brow, 96, lanehalf);
  int clab = labels[gcol];

  while (ct < nColTiles) {
    float8 acc = {};
    acc = __builtin_amdgcn_wmma_f32_16x16x32_f16(false, A0, false, B0,
                                                 (short)0, acc, false, false);
    acc = __builtin_amdgcn_wmma_f32_16x16x32_f16(false, A1, false, B1,
                                                 (short)0, acc, false, false);
    acc = __builtin_amdgcn_wmma_f32_16x16x32_f16(false, A2, false, B2,
                                                 (short)0, acc, false, false);
    acc = __builtin_amdgcn_wmma_f32_16x16x32_f16(false, A3, false, B3,
                                                 (short)0, acc, false, false);

    // Issue next tile's loads now; the scatter below hides their latency.
    const int ctn  = ct + HIST_WAVES;
    const int ctl  = (ctn < nColTiles) ? ctn : wave;   // clamped (no branch)
    const int ngcol = ctl * 16 + m;
    const _Float16* nbrow = xh + (size_t)ngcol * EMBED;
    half16 nB0 = load_frag(nbrow,  0, lanehalf);
    half16 nB1 = load_frag(nbrow, 32, lanehalf);
    half16 nB2 = load_frag(nbrow, 64, lanehalf);
    half16 nB3 = load_frag(nbrow, 96, lanehalf);
    const int nclab = labels[ngcol];

    // Scatter: triangle pulse hits exactly bins b and b+1 with (1-f, f).
    // Branchless: diagonal (grow == gcol) contributes 0 to both bins.
#pragma unroll
    for (int v = 0; v < 8; ++v) {
      const int r    = lanehalf * 8 + v;        // row within tile
      const int grow = rowBase + r;
      float d2 = 2.0f - 2.0f * acc[v];
      d2 = fminf(fmaxf(d2, 0.0f), 4.0f);
      float t = d2 * invDelta;
      int b = (int)t;
      b = (b > NBINS - 1) ? (NBINS - 1) : b;
      const float f    = t - (float)b;
      const float mask = (grow != gcol) ? 1.0f : 0.0f;
      float* h = (rl[v] == clab) ? &s_h[0][r][0] : &s_h[1][r][0];
      atomicAdd(h + b,     (1.0f - f) * mask);
      atomicAdd(h + b + 1, f * mask);
    }

    B0 = nB0; B1 = nB1; B2 = nB2; B3 = nB3;
    clab = nclab; gcol = ngcol;
    ct = ctn;
  }
  __syncthreads();

  for (int i = tid; i < 16 * LBINS; i += blockDim.x) {
    const int r = i / LBINS, b = i % LBINS;
    g_hpos[(size_t)(rowBase + r) * LBINS + b] = s_h[0][r][b];
    g_hneg[(size_t)(rowBase + r) * LBINS + b] = s_h[1][r][b];
  }
}

// ---------------- Phase 2: per-row AP + reduction ---------------------------
__global__ void fastap_ap_kernel(const float* __restrict__ g_hpos,
                                 const float* __restrict__ g_hneg,
                                 float* __restrict__ acc) {
  const int r = blockIdx.x * blockDim.x + threadIdx.x;
  if (r >= N_SAMPLES) return;
  float Hp = 0.0f, H = 0.0f, ap = 0.0f;
#pragma unroll
  for (int b = 0; b < LBINS; ++b) {
    const float hp = g_hpos[(size_t)r * LBINS + b];
    const float hn = g_hneg[(size_t)r * LBINS + b];
    Hp += hp;
    H  += hp + hn;
    if (H > 0.0f) ap += hp * Hp / H;
  }
  const float Np = Hp;                 // sum of pulses over bins == N_pos
  const float valid   = (Np > 0.0f) ? 1.0f : 0.0f;
  const float contrib = (Np > 0.0f) ? (ap / Np) : 0.0f;
  atomicAdd(&acc[0], contrib);
  atomicAdd(&acc[1], valid);
}

// ---------------- Phase 3: finalize -----------------------------------------
__global__ void fastap_final_kernel(const float* __restrict__ acc,
                                    float* __restrict__ out) {
  const float nv = (acc[1] > 0.0f) ? acc[1] : 1.0f;
  out[0] = 1.0f - acc[0] / nv;
}

// ---------------------------------------------------------------------------
extern "C" void kernel_launch(void* const* d_in, const int* in_sizes, int n_in,
                              void* d_out, int out_size, void* d_ws, size_t ws_size,
                              hipStream_t stream) {
  const float* x      = (const float*)d_in[0];
  const int*   labels = (const int*)d_in[1];
  float*       out    = (float*)d_out;

  char* ws = (char*)d_ws;
  _Float16* xh   = (_Float16*)ws;                               // 2 MB
  float*    hpos = (float*)(ws + (size_t)N_SAMPLES * EMBED * 2);
  float*    hneg = hpos + (size_t)N_SAMPLES * LBINS;
  float*    acc  = hneg + (size_t)N_SAMPLES * LBINS;            // 2 floats

  fastap_convert_kernel<<<(N_SAMPLES * EMBED + 255) / 256, 256, 0, stream>>>(x, xh, acc);
  fastap_hist_kernel<<<N_SAMPLES / 16, 32 * HIST_WAVES, 0, stream>>>(xh, labels, hpos, hneg);
  fastap_ap_kernel<<<(N_SAMPLES + 255) / 256, 256, 0, stream>>>(hpos, hneg, acc);
  fastap_final_kernel<<<1, 1, 0, stream>>>(acc, out);
}
